// MoE_20031727468767
// MI455X (gfx1250) — matrix operand order repaired and checked by reading, not compiled, not run on hardware
//
#include <hip/hip_runtime.h>
#include <cstdint>

typedef __attribute__((ext_vector_type(16))) _Float16 v16h;
typedef __attribute__((ext_vector_type(8)))  _Float16 v8h;
typedef __attribute__((ext_vector_type(8)))  float    v8f;
typedef unsigned int u32x4 __attribute__((ext_vector_type(4)));
typedef int          i32x8 __attribute__((ext_vector_type(8)));
typedef int          i32x4 __attribute__((ext_vector_type(4)));

#define TOK   8192          // B*S
#define DIM   2048          // d_hidden
#define DEXP  1408          // d_expert (also shared-chunk size)
#define NEXP  8
#define DSHH  2816          // shared hidden total (2 chunks of 1408)

#define BM    64            // token tile
#define BN    128           // output-column tile
#define BK    32            // WMMA K step
#define A_LD  40            // 32 + 8 pad (halves)
#define W_LD  40            // 32 + 8 pad (halves) == TDM pad 16DW interval + 4DW amount
#define H_LD  1416          // 1408 + 8 pad (halves)

#if defined(__HIP_DEVICE_COMPILE__) && defined(__has_builtin)
#if __has_builtin(__builtin_amdgcn_tensor_load_to_lds) && __has_builtin(__builtin_amdgcn_s_wait_tensorcnt)
#define HAVE_TDM 1
#endif
#endif
#ifndef HAVE_TDM
#define HAVE_TDM 0
#endif

// ---------------- WMMA helpers ----------------

__device__ __forceinline__ v8f wmma16(v16h a, v16h b, v8f c) {
  return __builtin_amdgcn_wmma_f32_16x16x32_f16(
      false, a, false, b, (short)0, c, false, false);
}

// A fragment 16x32 (MxK): lane L -> M=L&15; K chunks [hb,hb+8) and [16+hb,16+hb+8), hb=(L>=16)*8
__device__ __forceinline__ v16h frag_a(const _Float16* p, int row0, int k0, int ld) {
  const int lane = threadIdx.x & 31;
  const int m  = row0 + (lane & 15);
  const int hb = (lane >> 4) * 8;
  const _Float16* q = p + m * ld + k0 + hb;
  v8h lo = *(const v8h*)q;
  v8h hi = *(const v8h*)(q + 16);
  v16h r;
#pragma unroll
  for (int i = 0; i < 8; ++i) { r[i] = lo[i]; r[i + 8] = hi[i]; }
  return r;
}

// B fragment from row-major [N][K] tile: lane L -> N=L&15; K chunks [kb,kb+8) and [16+kb,16+kb+8), kb=(L>=16)*8
__device__ __forceinline__ v16h frag_b(const _Float16* p, int n0, int k0, int ld) {
  const int lane = threadIdx.x & 31;
  const int n  = n0 + (lane & 15);
  const int kb = (lane >> 4) * 8;
  const _Float16* q = p + n * ld + k0 + kb;
  v8h lo = *(const v8h*)q;
  v8h hi = *(const v8h*)(q + 16);
  v16h r;
#pragma unroll
  for (int i = 0; i < 8; ++i) { r[i] = lo[i]; r[i + 8] = hi[i]; }
  return r;
}

__device__ __forceinline__ v8f vzero8() {
  v8f z = {0.f, 0.f, 0.f, 0.f, 0.f, 0.f, 0.f, 0.f};
  return z;
}

// fast sigmoid: single v_exp_f32 + v_rcp_f32 (both TRANS ops, co-issue with WMMA)
__device__ __forceinline__ float fast_sigmoid(float v) {
  return __builtin_amdgcn_rcpf(1.f + __expf(-v));
}

__device__ __forceinline__ uint32_t lds_addr_of(const void* p) {
  // low 32 bits of a flat shared-aperture address are the LDS byte offset
  return (uint32_t)(uintptr_t)p;
}

// ---------------- Tensor Data Mover: 2D fp16 tile -> padded LDS ----------------
// tile: BK halves (K) x rows (N); global row stride ldHalves; LDS rows padded 32->40 halves.
__device__ __forceinline__ void tdm_load_tile_f16(uint32_t ldsAddr, const _Float16* gptr,
                                                  int rows, int ldHalves) {
#if HAVE_TDM
  const uint64_t ga = (uint64_t)(uintptr_t)gptr;
  u32x4 g0;
  g0[0] = 1u;                                   // count=1 valid descriptor
  g0[1] = ldsAddr;                              // lds_addr [63:32]
  g0[2] = (uint32_t)ga;                         // global_addr[31:0]  -> bits [95:64]
  g0[3] = (uint32_t)(ga >> 32) | (2u << 30);    // global_addr[56:32] -> [120:96]; type=2
  i32x8 g1;
  g1[0] = (int)((1u << 16)                      // data_size = 2 bytes
              | (1u << 20)                      // pad_enable
              | (3u << 22)                      // pad_interval: 16 DWORDs (= BK halves)
              | (3u << 25));                    // pad_amount:   4 DWORDs (= 8 halves)
  g1[1] = (int)(((uint32_t)BK & 0xFFFFu) << 16);    // tensor_dim0[15:0] -> [63:48]
  g1[2] = (int)(((uint32_t)rows & 0xFFFFu) << 16);  // tensor_dim1[15:0] -> [95:80]
  g1[3] = (int)(((uint32_t)BK & 0xFFFFu) << 16);    // tile_dim0 -> [127:112]
  g1[4] = (int)((uint32_t)rows & 0xFFFFu);          // tile_dim1 -> [143:128]; tile_dim2=0
  g1[5] = (int)(uint32_t)ldHalves;                  // tensor_dim0_stride[31:0]
  g1[6] = 0;
  g1[7] = 0;
  i32x4 z4 = {0, 0, 0, 0};
  i32x8 z8 = {0, 0, 0, 0, 0, 0, 0, 0};
  __builtin_amdgcn_tensor_load_to_lds(g0, g1, z4, z4, z8, 0);
#else
  (void)ldsAddr; (void)gptr; (void)rows; (void)ldHalves;
#endif
}

__device__ __forceinline__ void tdm_wait0() {
#if HAVE_TDM
  __builtin_amdgcn_s_wait_tensorcnt(0);
#endif
}

// VALU fallback staging (also used for host-pass parsing)
__device__ __forceinline__ void stage_w(_Float16* dst, const _Float16* src, int kk, int ld) {
  const int wRow = threadIdx.x >> 1, wSeg = threadIdx.x & 1;
  const _Float16* p = src + (size_t)wRow * ld + kk + wSeg * 16;
  *(v8h*)&dst[wRow * W_LD + wSeg * 16]     = ((const v8h*)p)[0];
  *(v8h*)&dst[wRow * W_LD + wSeg * 16 + 8] = ((const v8h*)p)[1];
}

// ---------------- gating: softmax + top-2 + compaction ----------------

__global__ __launch_bounds__(256)
void moe_gate(const float* __restrict__ x, const float* __restrict__ Wg,
              int* __restrict__ counts, int* __restrict__ tokIdx, float* __restrict__ tokW)
{
  const int lane = threadIdx.x & 31;
  const int wave = threadIdx.x >> 5;
  const int t = blockIdx.x * 8 + wave;

  float acc[NEXP];
#pragma unroll
  for (int e = 0; e < NEXP; ++e) acc[e] = 0.f;
  const float* xr = x + (size_t)t * DIM;
  for (int k = lane; k < DIM; k += 32) {
    float xv = xr[k];
#pragma unroll
    for (int e = 0; e < NEXP; ++e) acc[e] += xv * Wg[e * DIM + k];
  }
#pragma unroll
  for (int off = 16; off > 0; off >>= 1) {
#pragma unroll
    for (int e = 0; e < NEXP; ++e) acc[e] += __shfl_xor(acc[e], off, 32);
  }
  if (lane == 0) {
    float mx = acc[0];
#pragma unroll
    for (int e = 1; e < NEXP; ++e) mx = fmaxf(mx, acc[e]);
    float den = 0.f, sc[NEXP];
#pragma unroll
    for (int e = 0; e < NEXP; ++e) { sc[e] = __expf(acc[e] - mx); den += sc[e]; }
    float inv = 1.f / den;
#pragma unroll
    for (int e = 0; e < NEXP; ++e) sc[e] *= inv;
    int i0 = 0;
#pragma unroll
    for (int e = 1; e < NEXP; ++e) if (sc[e] > sc[i0]) i0 = e;
    int i1 = (i0 == 0) ? 1 : 0;
#pragma unroll
    for (int e = 0; e < NEXP; ++e) if (e != i0 && sc[e] > sc[i1]) i1 = e;
    int s0 = atomicAdd(&counts[i0], 1);
    tokIdx[i0 * TOK + s0] = t;  tokW[i0 * TOK + s0] = sc[i0];
    int s1 = atomicAdd(&counts[i1], 1);
    tokIdx[i1 * TOK + s1] = t;  tokW[i1 * TOK + s1] = sc[i1];
  }
}

// ---------------- fused expert: TDM-fed gate/up GEMM -> SwiGLU in LDS -> down GEMM ----------------

__global__ __launch_bounds__(256)
void moe_experts(const float* __restrict__ x,
                 const _Float16* __restrict__ Wgate, const _Float16* __restrict__ Wup,
                 const _Float16* __restrict__ Wdown,
                 const _Float16* __restrict__ Wsgate, const _Float16* __restrict__ Wsup,
                 const _Float16* __restrict__ Wsdown,
                 const int* __restrict__ counts, const int* __restrict__ tokIdx,
                 const float* __restrict__ tokW, float* __restrict__ out)
{
  __shared__ __align__(16) _Float16 hS[BM * H_LD];       // fused SwiGLU activations
  __shared__ __align__(16) _Float16 aS[2][BM * A_LD];    // double-buffered A (x fp32->fp16)
  __shared__ __align__(16) _Float16 wgS[2][BN * W_LD];   // double-buffered gate/down weights
  __shared__ __align__(16) _Float16 wuS[2][BN * W_LD];   // double-buffered up weights
  __shared__ int   tIdxS[BM];
  __shared__ float tWS[BM];

  const int job   = blockIdx.y;          // 0..7 routed experts, 8..9 shared chunks
  const int tile0 = blockIdx.x * BM;
  const bool routed = (job < NEXP);

  int rows = BM;
  const _Float16 *WgB, *WuB, *WdB;
  int ldW;
  if (routed) {
    const int e   = job;
    const int cnt = counts[e];
    if (tile0 >= cnt) return;            // uniform across block
    rows = min(BM, cnt - tile0);
    WgB = Wgate + (size_t)e * DEXP * DIM;
    WuB = Wup   + (size_t)e * DEXP * DIM;
    WdB = Wdown + (size_t)e * DIM * DEXP;
    ldW = DEXP;
  } else {
    const int c = job - NEXP;            // shared-expert hidden chunk 0/1
    WgB = Wsgate + (size_t)c * DEXP * DIM;
    WuB = Wsup   + (size_t)c * DEXP * DIM;
    WdB = Wsdown + (size_t)c * DEXP;     // column offset into [DIM, DSHH]
    ldW = DSHH;
  }

  const int tid = threadIdx.x;
  if (tid < BM) {
    if (routed) {
      const int cnt = counts[job];
      const int idx = tile0 + tid;
      if (idx < cnt) { tIdxS[tid] = tokIdx[job * TOK + idx]; tWS[tid] = tokW[job * TOK + idx]; }
      else           { tIdxS[tid] = tokIdx[job * TOK + tile0]; tWS[tid] = 0.f; }
    } else {
      tIdxS[tid] = tile0 + tid; tWS[tid] = 1.f;
    }
  }
  __syncthreads();

  const int lane   = tid & 31;
  const int waveId = tid >> 5;
  const int wm = waveId & 1;             // 2 M-blocks of 32 tokens
  const int wn = waveId >> 1;            // 4 N-blocks of 32 cols
  const int nIn   = lane & 15;
  const int hiAdd = (lane >> 4) * 8;
  const bool issuer = (waveId == 0);     // one wave drives the TDM

  const int aRow = tid >> 2, aSeg = tid & 3;   // A stage: 64 rows x 4 segs of 8

  auto stageA = [&](int buf, int kk) {
    const float* xp = x + (size_t)tIdxS[aRow] * DIM + kk + aSeg * 8;
    float4 f0 = ((const float4*)xp)[0];
    float4 f1 = ((const float4*)xp)[1];
    v8h h8;
    h8[0] = (_Float16)f0.x; h8[1] = (_Float16)f0.y; h8[2] = (_Float16)f0.z; h8[3] = (_Float16)f0.w;
    h8[4] = (_Float16)f1.x; h8[5] = (_Float16)f1.y; h8[6] = (_Float16)f1.z; h8[7] = (_Float16)f1.w;
    *(v8h*)&aS[buf][aRow * A_LD + aSeg * 8] = h8;
  };

  // ---------------- phase 1: h = silu(x@Wg^T) * (x@Wu^T) * w -> LDS ----------------
  const int nK1 = DIM / BK;              // 64
  for (int hT = 0; hT < DEXP / BN; ++hT) {
    v8f gacc[2][2], uacc[2][2];
#pragma unroll
    for (int i = 0; i < 2; ++i)
#pragma unroll
      for (int j = 0; j < 2; ++j) { gacc[i][j] = vzero8(); uacc[i][j] = vzero8(); }

    const _Float16* wgTile = WgB + (size_t)(hT * BN) * DIM;
    const _Float16* wuTile = WuB + (size_t)(hT * BN) * DIM;

    // prologue: fill buffer 0
    stageA(0, 0);
#if HAVE_TDM
    if (issuer) {
      tdm_load_tile_f16(lds_addr_of(&wgS[0][0]), wgTile, BN, DIM);
      tdm_load_tile_f16(lds_addr_of(&wuS[0][0]), wuTile, BN, DIM);
    }
#else
    stage_w(wgS[0], wgTile, 0, DIM);
    stage_w(wuS[0], wuTile, 0, DIM);
#endif

    for (int kt = 0; kt < nK1; ++kt) {
      const int cur = kt & 1;
      if (issuer) tdm_wait0();           // buffer `cur` complete (TDM path)
      __syncthreads();                   // publish buffer `cur`; all done with `cur^1`
      if (kt + 1 < nK1) {                // prefetch next slice into the other buffer
        stageA(cur ^ 1, (kt + 1) * BK);
#if HAVE_TDM
        if (issuer) {
          tdm_load_tile_f16(lds_addr_of(&wgS[cur ^ 1][0]), wgTile + (kt + 1) * BK, BN, DIM);
          tdm_load_tile_f16(lds_addr_of(&wuS[cur ^ 1][0]), wuTile + (kt + 1) * BK, BN, DIM);
        }
#else
        stage_w(wgS[cur ^ 1], wgTile, (kt + 1) * BK, DIM);
        stage_w(wuS[cur ^ 1], wuTile, (kt + 1) * BK, DIM);
#endif
      }

      v16h a0 = frag_a(aS[cur], wm * 32,      0, A_LD);
      v16h a1 = frag_a(aS[cur], wm * 32 + 16, 0, A_LD);
      v16h g0 = frag_b(wgS[cur], wn * 32,      0, W_LD);
      v16h g1 = frag_b(wgS[cur], wn * 32 + 16, 0, W_LD);
      v16h u0 = frag_b(wuS[cur], wn * 32,      0, W_LD);
      v16h u1 = frag_b(wuS[cur], wn * 32 + 16, 0, W_LD);

      gacc[0][0] = wmma16(a0, g0, gacc[0][0]);
      gacc[0][1] = wmma16(a0, g1, gacc[0][1]);
      gacc[1][0] = wmma16(a1, g0, gacc[1][0]);
      gacc[1][1] = wmma16(a1, g1, gacc[1][1]);
      uacc[0][0] = wmma16(a0, u0, uacc[0][0]);
      uacc[0][1] = wmma16(a0, u1, uacc[0][1]);
      uacc[1][0] = wmma16(a1, u0, uacc[1][0]);
      uacc[1][1] = wmma16(a1, u1, uacc[1][1]);
    }

    // SwiGLU + route weight, scatter into hS
#pragma unroll
    for (int mf = 0; mf < 2; ++mf)
#pragma unroll
      for (int nf = 0; nf < 2; ++nf) {
        v8f g = gacc[mf][nf], u = uacc[mf][nf];
#pragma unroll
        for (int r = 0; r < 8; ++r) {
          const int m  = wm * 32 + mf * 16 + hiAdd + r;
          const float gv = g[r], uv = u[r];
          const float hv = gv * fast_sigmoid(gv) * uv * tWS[m];
          hS[m * H_LD + hT * BN + wn * 32 + nf * 16 + nIn] = (_Float16)hv;
        }
      }
  }
  __syncthreads();   // hS fully written before down-projection reads it

  // ---------------- phase 2: out += h @ Wd^T (A straight from LDS) ----------------
  const int nK2 = DEXP / BK;             // 44
  for (int dT = 0; dT < DIM / BN; ++dT) {
    v8f acc[2][2];
#pragma unroll
    for (int i = 0; i < 2; ++i)
#pragma unroll
      for (int j = 0; j < 2; ++j) acc[i][j] = vzero8();

    const _Float16* wdTile = WdB + (size_t)(dT * BN) * ldW;

#if HAVE_TDM
    if (issuer) tdm_load_tile_f16(lds_addr_of(&wgS[0][0]), wdTile, BN, ldW);
#else
    stage_w(wgS[0], wdTile, 0, ldW);
#endif

    for (int kt = 0; kt < nK2; ++kt) {
      const int cur = kt & 1;
      if (issuer) tdm_wait0();
      __syncthreads();
      if (kt + 1 < nK2) {
#if HAVE_TDM
        if (issuer)
          tdm_load_tile_f16(lds_addr_of(&wgS[cur ^ 1][0]), wdTile + (kt + 1) * BK, BN, ldW);
#else
        stage_w(wgS[cur ^ 1], wdTile, (kt + 1) * BK, ldW);
#endif
      }

      v16h a0 = frag_a(hS, wm * 32,      kt * BK, H_LD);
      v16h a1 = frag_a(hS, wm * 32 + 16, kt * BK, H_LD);
      v16h d0 = frag_b(wgS[cur], wn * 32,      0, W_LD);
      v16h d1 = frag_b(wgS[cur], wn * 32 + 16, 0, W_LD);

      acc[0][0] = wmma16(a0, d0, acc[0][0]);
      acc[0][1] = wmma16(a0, d1, acc[0][1]);
      acc[1][0] = wmma16(a1, d0, acc[1][0]);
      acc[1][1] = wmma16(a1, d1, acc[1][1]);
    }

#pragma unroll
    for (int mf = 0; mf < 2; ++mf)
#pragma unroll
      for (int nf = 0; nf < 2; ++nf) {
#pragma unroll
        for (int r = 0; r < 8; ++r) {
          const int m = wm * 32 + mf * 16 + hiAdd + r;
          if (m < rows) {
            const int col = dT * BN + wn * 32 + nf * 16 + nIn;
            atomicAdd(out + (size_t)tIdxS[m] * DIM + col, acc[mf][nf][r]);
          }
        }
      }
  }
}

extern "C" void kernel_launch(void* const* d_in, const int* in_sizes, int n_in,
                              void* d_out, int out_size, void* d_ws, size_t ws_size,
                              hipStream_t stream) {
  (void)in_sizes; (void)n_in; (void)out_size; (void)ws_size;
  const float*    x       = (const float*)d_in[0];
  const float*    Wg      = (const float*)d_in[1];
  const _Float16* W_gate  = (const _Float16*)d_in[2];
  const _Float16* W_up    = (const _Float16*)d_in[3];
  const _Float16* W_down  = (const _Float16*)d_in[4];
  const _Float16* Ws_gate = (const _Float16*)d_in[5];
  const _Float16* Ws_up   = (const _Float16*)d_in[6];
  const _Float16* Ws_down = (const _Float16*)d_in[7];
  float* out = (float*)d_out;

  int*   counts = (int*)d_ws;
  int*   tokIdx = counts + 32;               // E*T ints
  float* tokW   = (float*)(tokIdx + NEXP * TOK);

  (void)hipMemsetAsync(d_out, 0, sizeof(float) * (size_t)TOK * DIM, stream);
  (void)hipMemsetAsync(counts, 0, 32 * sizeof(int), stream);

  moe_gate<<<TOK / 8, 256, 0, stream>>>(x, Wg, counts, tokIdx, tokW);
  moe_experts<<<dim3(TOK / BM, NEXP + 2), 256, 0, stream>>>(
      x, W_gate, W_up, W_down, Ws_gate, Ws_up, Ws_down, counts, tokIdx, tokW, out);
}
